// Node_Layer_Embedding_Model_72773925863768
// MI455X (gfx1250) — matrix operand order repaired and checked by reading, not compiled, not run on hardware
//
#include <hip/hip_runtime.h>

// ---------------------------------------------------------------------------
// Types for CDNA5 WMMA (wave32): bf16 A/B fragments (16 bf16 = 8 VGPRs),
// f32 accumulators (8 f32 = 8 VGPRs).
// ---------------------------------------------------------------------------
typedef __attribute__((ext_vector_type(16))) __bf16        bf16x16;
typedef __attribute__((ext_vector_type(8)))  float         v8f;
typedef __attribute__((ext_vector_type(4)))  unsigned int  u32x4;

union BFrag {
    u32x4   u[2];   // two 16-byte chunks (8 bf16 each)
    bf16x16 v;      // WMMA operand view
};

__device__ __forceinline__ unsigned short f2bf(float f) {
    unsigned u = __builtin_bit_cast(unsigned, f);
    u += 0x7FFFu + ((u >> 16) & 1u);          // round-to-nearest-even
    return (unsigned short)(u >> 16);
}
__device__ __forceinline__ float bf2f(unsigned short h) {
    unsigned u = ((unsigned)h) << 16;
    return __builtin_bit_cast(float, u);
}

// ---------------------------------------------------------------------------
// fp32 -> bf16 conversion (grid-stride)
// ---------------------------------------------------------------------------
__global__ void cvt_f32_bf16(const float* __restrict__ s,
                             unsigned short* __restrict__ d, size_t n) {
    size_t i  = (size_t)blockIdx.x * blockDim.x + threadIdx.x;
    size_t st = (size_t)gridDim.x * blockDim.x;
    for (; i < n; i += st) d[i] = f2bf(s[i]);
}

// fp32 [rows,16] -> bf16 [rows,64] zero-padded (so K is a multiple of 64)
__global__ void cvt_pad16to64(const float* __restrict__ s,
                              unsigned short* __restrict__ d, int rows) {
    int i = blockIdx.x * blockDim.x + threadIdx.x;
    int total = rows * 64;
    if (i >= total) return;
    int r = i >> 6, c = i & 63;
    d[i] = (c < 16) ? f2bf(s[r * 16 + c]) : (unsigned short)0;
}

// ---------------------------------------------------------------------------
// GEMM: Y[M,fout] = act(X[M,fin] * W[fout,fin]^T + bias), bf16 in/out, f32 acc.
//
// Block = 256 threads = 8 waves (wave32). Workgroup tile 128(M) x 128(N).
// Wave (wave&3 -> M, wave>>2 -> N) owns a 32x64 tile = 2x4 WMMA accumulators.
//
// K-stage = 64: A 128x64 and B 128x64 bf16 tiles are async-DMA'd into LDS
// (GLOBAL_LOAD_ASYNC_TO_LDS_B128, ASYNCcnt, 8 ops/thread/stage),
// double-buffered; the loads for stage i+1 overlap the 16 WMMAs of stage i.
// Steady state: s_wait_asynccnt 8 retires stage i while stage i+1 stays in
// flight. Last stage peeled -> no conditionals in the hot loop.
//
// LDS rows use a 72-element pitch (144B, 16B aligned) to spread ds_load_b128
// fragment reads across banks.
//
// Fragment layouts per CDNA5 ISA (16-bit, wave32):
//   A 16x32: lane L -> row M = L&15; lanes 0-15 hold K=0..7,16..23,
//            lanes 16-31 hold K=8..15,24..31 (2 bf16 per VGPR).
//   B 32x16: lane L -> col N = L&15; lanes 0-15 hold K=0..7,16..23,
//            lanes 16-31 hold K=8..15,24..31.
//   C/D 16x16: VGPR r -> row r (lanes 0-15) / row r+8 (lanes 16-31), col=L&15.
// ---------------------------------------------------------------------------
#define TM     128
#define TN     128
#define KSTG   64
#define PITCH  72   // LDS row pitch in bf16 elements (144 bytes)

__global__ __launch_bounds__(256)
void gemm_bf16_lds(const unsigned short* __restrict__ X,
                   const unsigned short* __restrict__ W,
                   const float* __restrict__ bias,
                   unsigned short* __restrict__ Y,
                   int fin, int fout, int relu) {
    __shared__ unsigned short smA[2][TM * PITCH];
    __shared__ unsigned short smB[2][TN * PITCH];

    const int tid  = threadIdx.x;
    const int lane = tid & 31;
    const int wave = tid >> 5;
    const int l16  = lane & 15;
    const bool hi  = lane >= 16;

    const int wm   = (wave & 3) * 32;   // wave M offset in block tile
    const int wn   = (wave >> 2) * 64;  // wave N offset in block tile
    const int mBlk = blockIdx.x * TM;
    const int nBlk = blockIdx.y * TN;

    // Cooperative stage mapping: 128 rows x 64 cols per tile = 1024 chunks of
    // 16B; 4 chunks per array per thread (rows tid>>3 + {0,32,64,96}).
    const int row0 = tid >> 3;
    const int colE = (tid & 7) * 8;

    const unsigned short* gA[4];
    const unsigned short* gB[4];
#pragma unroll
    for (int j = 0; j < 4; ++j) {
        gA[j] = X + (size_t)(mBlk + row0 + 32 * j) * fin + colE;
        gB[j] = W + (size_t)(nBlk + row0 + 32 * j) * fin + colE;
    }

    // Issue one K-stage (64 wide) of async tile loads into LDS buffer `buf`.
    auto stage = [&](int buf, int k) {
        unsigned la0 = (unsigned)(size_t)&smA[buf][(row0 +  0) * PITCH + colE];
        unsigned la1 = (unsigned)(size_t)&smA[buf][(row0 + 32) * PITCH + colE];
        unsigned la2 = (unsigned)(size_t)&smA[buf][(row0 + 64) * PITCH + colE];
        unsigned la3 = (unsigned)(size_t)&smA[buf][(row0 + 96) * PITCH + colE];
        unsigned lb0 = (unsigned)(size_t)&smB[buf][(row0 +  0) * PITCH + colE];
        unsigned lb1 = (unsigned)(size_t)&smB[buf][(row0 + 32) * PITCH + colE];
        unsigned lb2 = (unsigned)(size_t)&smB[buf][(row0 + 64) * PITCH + colE];
        unsigned lb3 = (unsigned)(size_t)&smB[buf][(row0 + 96) * PITCH + colE];
        asm volatile(
            "global_load_async_to_lds_b128 %0, %8, off\n\t"
            "global_load_async_to_lds_b128 %1, %9, off\n\t"
            "global_load_async_to_lds_b128 %2, %10, off\n\t"
            "global_load_async_to_lds_b128 %3, %11, off\n\t"
            "global_load_async_to_lds_b128 %4, %12, off\n\t"
            "global_load_async_to_lds_b128 %5, %13, off\n\t"
            "global_load_async_to_lds_b128 %6, %14, off\n\t"
            "global_load_async_to_lds_b128 %7, %15, off"
            :: "v"(la0), "v"(la1), "v"(la2), "v"(la3),
               "v"(lb0), "v"(lb1), "v"(lb2), "v"(lb3),
               "v"(gA[0] + k), "v"(gA[1] + k), "v"(gA[2] + k), "v"(gA[3] + k),
               "v"(gB[0] + k), "v"(gB[1] + k), "v"(gB[2] + k), "v"(gB[3] + k)
            : "memory");
    };

    v8f acc[2][4] = {};
    const int aSel = hi ? 8 : 0;    // A: hi lanes hold K=8..15 / 24..31
    const int bSel = hi ? 8 : 0;    // B: hi lanes hold K=8..15 / 24..31

    // Consume one staged buffer: two 32-wide K sub-steps, 16 WMMAs.
    auto compute = [&](int buf) {
#pragma unroll
        for (int ks = 0; ks < KSTG; ks += 32) {
            BFrag a[2], b[4];
#pragma unroll
            for (int mi = 0; mi < 2; ++mi) {
                const unsigned short* p =
                    &smA[buf][(wm + mi * 16 + l16) * PITCH + ks + aSel];
                a[mi].u[0] = *(const u32x4*)p;
                a[mi].u[1] = *(const u32x4*)(p + 16);
            }
#pragma unroll
            for (int ni = 0; ni < 4; ++ni) {
                const unsigned short* p =
                    &smB[buf][(wn + ni * 16 + l16) * PITCH + ks + bSel];
                b[ni].u[0] = *(const u32x4*)p;
                b[ni].u[1] = *(const u32x4*)(p + 16);
            }
#pragma unroll
            for (int mi = 0; mi < 2; ++mi)
#pragma unroll
                for (int ni = 0; ni < 4; ++ni)
                    acc[mi][ni] = __builtin_amdgcn_wmma_f32_16x16x32_bf16(
                        false, a[mi].v, false, b[ni].v, (short)0,
                        acc[mi][ni], false, false);
        }
    };

    const int nStages = fin >> 6;   // all fin are multiples of 64

    stage(0, 0);
    for (int i = 0; i + 1 < nStages; ++i) {
        stage((i + 1) & 1, (i + 1) * KSTG);
        asm volatile("s_wait_asynccnt 8" ::: "memory");
        __syncthreads();            // stage i fully in LDS
        compute(i & 1);
        __syncthreads();            // all waves done with buf before restage
    }
    asm volatile("s_wait_asynccnt 0" ::: "memory");
    __syncthreads();
    compute((nStages - 1) & 1);     // peeled last stage: no conditionals in loop

    const int rowSel = hi ? 8 : 0;
#pragma unroll
    for (int mi = 0; mi < 2; ++mi) {
#pragma unroll
        for (int ni = 0; ni < 4; ++ni) {
            const int n  = nBlk + wn + ni * 16 + l16;
            const float bv = bias[n];
            const int r0 = mBlk + wm + mi * 16 + rowSel;
            v8f c = acc[mi][ni];
#pragma unroll
            for (int r = 0; r < 8; ++r) {
                float v = c[r] + bv;
                if (relu) v = fmaxf(v, 0.0f);
                Y[(size_t)(r0 + r) * fout + n] = f2bf(v);
            }
        }
    }
}

// ---------------------------------------------------------------------------
// One wave per row: mean over 256 bf16 values -> f32
// ---------------------------------------------------------------------------
__global__ __launch_bounds__(256)
void row_mean_bf16(const unsigned short* __restrict__ Ybf,
                   float* __restrict__ mean) {
    int lane = threadIdx.x & 31, wave = threadIdx.x >> 5;
    int row = blockIdx.x * 8 + wave;
    const unsigned short* p = Ybf + (size_t)row * 256 + lane * 8;
    float s = 0.f;
#pragma unroll
    for (int j = 0; j < 8; ++j) s += bf2f(p[j]);
    s += __shfl_xor(s, 16); s += __shfl_xor(s, 8); s += __shfl_xor(s, 4);
    s += __shfl_xor(s, 2);  s += __shfl_xor(s, 1);
    if (lane == 0) mean[row] = s * (1.0f / 256.0f);
}

// One wave per row: out[b] = mean[b] * dot(layer_y[b,:], wreg)
__global__ __launch_bounds__(256)
void final_dot(const unsigned short* __restrict__ Lbf,
               const float* __restrict__ wreg,
               const float* __restrict__ mean,
               float* __restrict__ out) {
    int lane = threadIdx.x & 31, wave = threadIdx.x >> 5;
    int row = blockIdx.x * 8 + wave;
    const unsigned short* p = Lbf + (size_t)row * 256 + lane * 8;
    float s = 0.f;
#pragma unroll
    for (int j = 0; j < 8; ++j) s += bf2f(p[j]) * wreg[lane * 8 + j];
    s += __shfl_xor(s, 16); s += __shfl_xor(s, 8); s += __shfl_xor(s, 4);
    s += __shfl_xor(s, 2);  s += __shfl_xor(s, 1);
    if (lane == 0) out[row] = mean[row] * s;
}

// ---------------------------------------------------------------------------
// Host-side orchestration
// ---------------------------------------------------------------------------
extern "C" void kernel_launch(void* const* d_in, const int* in_sizes, int n_in,
                              void* d_out, int out_size, void* d_ws, size_t ws_size,
                              hipStream_t stream) {
    (void)in_sizes; (void)n_in; (void)out_size; (void)ws_size;
    constexpr int B = 32768;

    const float* node_x  = (const float*)d_in[0];
    const float* layer_x = (const float*)d_in[1];
    const float* nW[5] = {(const float*)d_in[2], (const float*)d_in[4],
                          (const float*)d_in[6], (const float*)d_in[8],
                          (const float*)d_in[10]};
    const float* nb[5] = {(const float*)d_in[3], (const float*)d_in[5],
                          (const float*)d_in[7], (const float*)d_in[9],
                          (const float*)d_in[11]};
    const float* lW[5] = {(const float*)d_in[12], (const float*)d_in[14],
                          (const float*)d_in[16], (const float*)d_in[18],
                          (const float*)d_in[20]};
    const float* lb[5] = {(const float*)d_in[13], (const float*)d_in[15],
                          (const float*)d_in[17], (const float*)d_in[19],
                          (const float*)d_in[21]};
    const float* wreg = (const float*)d_in[22];
    float* out = (float*)d_out;

    const int nDims[6] = {256, 256, 512, 1024, 512, 256};
    const int lDims[6] = {64, 256, 512, 1024, 512, 256};  // K of layer L1 padded to 64

    // workspace layout (bf16 element offsets; all sizes multiples of 8 -> 16B aligned)
    unsigned short* ws = (unsigned short*)d_ws;
    size_t off = 0;
    unsigned short* wbN[5];
    unsigned short* wbL[5];
    for (int i = 0; i < 5; ++i) { wbN[i] = ws + off; off += (size_t)nDims[i + 1] * nDims[i]; }
    for (int i = 0; i < 5; ++i) { wbL[i] = ws + off; off += (size_t)lDims[i + 1] * lDims[i]; }
    unsigned short* buf0 = ws + off; off += (size_t)B * 1024;
    unsigned short* buf1 = ws + off; off += (size_t)B * 1024;
    float* meanb = (float*)(ws + off);

    auto cvt = [&](const float* s, unsigned short* d, size_t n) {
        int blocks = (int)((n + 255) / 256);
        if (blocks > 4096) blocks = 4096;
        cvt_f32_bf16<<<blocks, 256, 0, stream>>>(s, d, n);
    };
    auto gemm = [&](const unsigned short* Xb, const unsigned short* Wb,
                    const float* bias, unsigned short* Yb,
                    int fin, int fout, int relu) {
        dim3 g(B / TM, fout / TN);
        gemm_bf16_lds<<<g, 256, 0, stream>>>(Xb, Wb, bias, Yb, fin, fout, relu);
    };

    // --- weight conversion (deterministic, every call) ---
    for (int i = 0; i < 5; ++i) cvt(nW[i], wbN[i], (size_t)nDims[i + 1] * nDims[i]);
    cvt_pad16to64<<<(256 * 64 + 255) / 256, 256, 0, stream>>>(lW[0], wbL[0], 256);
    for (int i = 1; i < 5; ++i) cvt(lW[i], wbL[i], (size_t)lDims[i + 1] * lDims[i]);

    // --- node tower ---
    cvt(node_x, buf0, (size_t)B * 256);
    gemm(buf0, wbN[0], nb[0], buf1, 256, 256, 1);
    gemm(buf1, wbN[1], nb[1], buf0, 256, 512, 1);
    gemm(buf0, wbN[2], nb[2], buf1, 512, 1024, 1);
    gemm(buf1, wbN[3], nb[3], buf0, 1024, 512, 1);
    gemm(buf0, wbN[4], nb[4], buf1, 512, 256, 0);
    row_mean_bf16<<<B / 8, 256, 0, stream>>>(buf1, meanb);

    // --- layer tower (reuses buffers after mean reduction) ---
    cvt_pad16to64<<<(B * 64 + 255) / 256, 256, 0, stream>>>(layer_x, buf0, B);
    gemm(buf0, wbL[0], lb[0], buf1, 64, 256, 1);
    gemm(buf1, wbL[1], lb[1], buf0, 256, 512, 1);
    gemm(buf0, wbL[2], lb[2], buf1, 512, 1024, 1);
    gemm(buf1, wbL[3], lb[3], buf0, 1024, 512, 1);
    gemm(buf0, wbL[4], lb[4], buf1, 512, 256, 0);

    // --- collapsed head: out[b] = mean(node_y[b]) * (layer_y[b] . wreg) ---
    final_dot<<<B / 8, 256, 0, stream>>>(buf1, wreg, meanb, out);
}
